// ObjectModel_40776419508248
// MI455X (gfx1250) — matrix operand order repaired and checked by reading, not compiled, not run on hardware
//
#include <hip/hip_runtime.h>
#include <hip/hip_bf16.h>
#include <math.h>

// ---------------------------------------------------------------------------
// Problem constants
// ---------------------------------------------------------------------------
#define B_   64
#define S1_  512
#define S2_  512
#define H_   768
#define DW_  200
#define DWP  224   // DW padded to multiple of 32 (serves both K-pad and N-pad)

typedef __attribute__((ext_vector_type(16))) __bf16 v16bf;
typedef __attribute__((ext_vector_type(8)))  __bf16 v8bf;
typedef __attribute__((ext_vector_type(8)))  float  v8f;

// ---------------------------------------------------------------------------
// Workspace layout (all sizes multiples of 256 bytes)
// ---------------------------------------------------------------------------
static constexpr size_t SZ_A1   = (size_t)B_ * S1_ * H_  * 2;   // x1 bf16
static constexpr size_t SZ_K1   = (size_t)B_ * S2_ * H_  * 2;   // x2 bf16
static constexpr size_t SZ_X2T  = (size_t)B_ * H_  * S2_ * 2;   // x2^T bf16 [B][768][512]
static constexpr size_t SZ_A2   = (size_t)B_ * S1_ * DWP * 2;   // tt bf16 padded
static constexpr size_t SZ_K2   = (size_t)B_ * S2_ * DWP * 2;   // tt2 bf16 padded
static constexpr size_t SZ_T2T  = (size_t)B_ * DWP * S2_ * 2;   // tt2^T bf16 [B][224][512]
static constexpr size_t SZ_WBT  = (size_t)H_ * H_ * 2;          // W_bert^T bf16 [768][768]
static constexpr size_t SZ_WLB  = (size_t)H_ * 4;               // W_bert last row fp32
static constexpr size_t SZ_WWT  = (size_t)DWP * DWP * 2;        // W_wv^T bf16 [224][224]
static constexpr size_t SZ_WLW  = 1024;                         // W_wv last row fp32 [224]
static constexpr size_t SZ_Q1   = (size_t)B_ * S1_ * H_  * 2;   // Q_bert bf16
static constexpr size_t SZ_Q2   = (size_t)B_ * S1_ * DWP * 2;   // Q_wv bf16
static constexpr size_t SZ_S    = (size_t)B_ * S1_ * S2_ * 4;   // scores fp32 (reused)
static constexpr size_t SZ_P    = (size_t)B_ * S1_ * S2_ * 2;   // probs bf16 (reused)
static constexpr size_t SZ_V    = (size_t)B_ * S1_ * H_  * 4;   // v fp32 (reused; wv ld=224 fits)
static constexpr size_t SZ_X12  = (size_t)B_ * H_ * 4;
static constexpr size_t SZ_X12W = (size_t)B_ * DWP * 4;
static constexpr size_t SZ_X12WH= (size_t)B_ * H_ * 4;

static constexpr size_t OFF_A1    = 0;
static constexpr size_t OFF_K1    = OFF_A1    + SZ_A1;
static constexpr size_t OFF_X2T   = OFF_K1    + SZ_K1;
static constexpr size_t OFF_A2    = OFF_X2T   + SZ_X2T;
static constexpr size_t OFF_K2    = OFF_A2    + SZ_A2;
static constexpr size_t OFF_T2T   = OFF_K2    + SZ_K2;
static constexpr size_t OFF_WBT   = OFF_T2T   + SZ_T2T;
static constexpr size_t OFF_WLB   = OFF_WBT   + SZ_WBT;
static constexpr size_t OFF_WWT   = OFF_WLB   + SZ_WLB;
static constexpr size_t OFF_WLW   = OFF_WWT   + SZ_WWT;
static constexpr size_t OFF_Q1    = OFF_WLW   + SZ_WLW;
static constexpr size_t OFF_Q2    = OFF_Q1    + SZ_Q1;
static constexpr size_t OFF_S     = OFF_Q2    + SZ_Q2;
static constexpr size_t OFF_P     = OFF_S     + SZ_S;
static constexpr size_t OFF_V     = OFF_P     + SZ_P;
static constexpr size_t OFF_X12   = OFF_V     + SZ_V;
static constexpr size_t OFF_X12W  = OFF_X12   + SZ_X12;
static constexpr size_t OFF_X12WH = OFF_X12W  + SZ_X12W;

// ---------------------------------------------------------------------------
// WMMA fragment loaders (bf16, A row-major [M][K], B staged N-major Bt[n][k])
//   A 16x32 layout: lane m = lane%16, element e: K = (e/8)*16 + half*8 + e%8
//   B 32x16 layout: lane n = lane%16, element e: K = (e/8)*16 + half*8 + e%8
// Both reduce to two contiguous 16-byte loads per lane.
// ---------------------------------------------------------------------------
__device__ __forceinline__ v16bf load_a_frag(const __bf16* A, int lda, int row,
                                             int kb, int hl) {
    const __bf16* p = A + (size_t)row * lda + kb + hl * 8;
    v8bf lo = *(const v8bf*)(p);
    v8bf hi = *(const v8bf*)(p + 16);
    v16bf r;
#pragma unroll
    for (int i = 0; i < 8; ++i) { r[i] = lo[i]; r[i + 8] = hi[i]; }
    return r;
}

__device__ __forceinline__ v16bf load_b_frag(const __bf16* Bt, int ldb, int col,
                                             int kb, int hl) {
    const __bf16* p = Bt + (size_t)col * ldb + kb + hl * 8;
    v8bf lo = *(const v8bf*)(p);
    v8bf hi = *(const v8bf*)(p + 16);
    v16bf r;
#pragma unroll
    for (int i = 0; i < 8; ++i) { r[i] = lo[i]; r[i + 8] = hi[i]; }
    return r;
}

// ---------------------------------------------------------------------------
// Prep kernels
// ---------------------------------------------------------------------------
__global__ void pack_bf16_kernel(const float* __restrict__ src, __bf16* __restrict__ dst,
                                 long rows, int cols, int ldd) {
    long idx = (long)blockIdx.x * blockDim.x + threadIdx.x;
    if (idx >= rows * (long)ldd) return;
    long r = idx / ldd;
    int  c = (int)(idx % ldd);
    float v = (c < cols) ? src[r * cols + c] : 0.0f;
    dst[idx] = (__bf16)v;
}

__global__ void transpose_conv_kernel(const float* __restrict__ src, __bf16* __restrict__ dst,
                                      int T, int Dsrc, int Dpad) {
    // src [B][T][Dsrc] fp32  ->  dst [B][Dpad][T] bf16 (rows >= Dsrc zero)
    long idx = (long)blockIdx.x * blockDim.x + threadIdx.x;
    long total = (long)B_ * Dpad * T;
    if (idx >= total) return;
    int b = (int)(idx / ((long)Dpad * T));
    int r = (int)(idx % ((long)Dpad * T));
    int d = r / T, t = r % T;
    float v = (d < Dsrc) ? src[((size_t)b * T + t) * Dsrc + d] : 0.0f;
    dst[idx] = (__bf16)v;
}

__global__ void prep_w_kernel(const float* __restrict__ W, __bf16* __restrict__ WT,
                              float* __restrict__ wlast, int K, int N, int Kpad, int Npad) {
    // W [(K+1)][N] fp32 -> WT[n][k] bf16 [Npad][Kpad]; wlast[n] = W[K][n]
    long idx = (long)blockIdx.x * blockDim.x + threadIdx.x;
    if (idx >= (long)Npad * Kpad) return;
    int n = (int)(idx / Kpad), k = (int)(idx % Kpad);
    float v = (n < N && k < K) ? W[(size_t)k * N + n] : 0.0f;
    WT[idx] = (__bf16)v;
    if (k == 0) wlast[n] = (n < N) ? W[(size_t)K * N + n] : 0.0f;
}

// ---------------------------------------------------------------------------
// Register-blocked Q = X @ W (+ rank-1 y*wlast), bf16 out.
// One wave computes a (MB*16) x (NB*16) macro-tile: MB*NB WMMAs per K-step
// from only MB+NB fragment loads -> ~2.7x less L2 traffic than 1x1.
// ---------------------------------------------------------------------------
template <int MB, int NB>
__global__ void wmma_gemm_q_kernel(const __bf16* __restrict__ A, int lda,
                                   const __bf16* __restrict__ Bt, int ldb,
                                   __bf16* __restrict__ Out, int ldo,
                                   const float* __restrict__ y,
                                   const float* __restrict__ wlast,
                                   int Ntiles2, int K) {
    int wave = blockIdx.x * (blockDim.x >> 5) + (threadIdx.x >> 5);
    int lane = threadIdx.x & 31;
    int mt = wave / Ntiles2, nt = wave % Ntiles2;
    int m0 = mt * 16 * MB, n0 = nt * 16 * NB;
    int hl = lane >> 4, l16 = lane & 15;
    v8f c[MB][NB];
#pragma unroll
    for (int i = 0; i < MB; ++i)
#pragma unroll
        for (int j = 0; j < NB; ++j) c[i][j] = (v8f){};
    for (int kb = 0; kb < K; kb += 32) {
        v16bf a[MB], b[NB];
#pragma unroll
        for (int i = 0; i < MB; ++i) a[i] = load_a_frag(A, lda, m0 + i * 16 + l16, kb, hl);
#pragma unroll
        for (int j = 0; j < NB; ++j) b[j] = load_b_frag(Bt, ldb, n0 + j * 16 + l16, kb, hl);
#pragma unroll
        for (int i = 0; i < MB; ++i)
#pragma unroll
            for (int j = 0; j < NB; ++j)
                c[i][j] = __builtin_amdgcn_wmma_f32_16x16x32_bf16(
                    false, a[i], false, b[j], (short)0, c[i][j], false, false);
    }
#pragma unroll
    for (int i = 0; i < MB; ++i)
#pragma unroll
        for (int j = 0; j < NB; ++j) {
            int n = n0 + j * 16 + l16;
            float wl = wlast[n];
#pragma unroll
            for (int r = 0; r < 8; ++r) {
                int row = m0 + i * 16 + r + hl * 8;
                float val = c[i][j][r] + y[row] * wl;
                Out[(size_t)row * ldo + n] = (__bf16)val;
            }
        }
}

// ---------------------------------------------------------------------------
// Register-blocked batched C = A @ Bt^T (Bt N-major), fp32 out.
// Used for scores (Bt = Keys row-major) and P@V (Bt = V transposed).
// ---------------------------------------------------------------------------
template <int MB, int NB>
__global__ void wmma_gemm_f32_kernel(const __bf16* __restrict__ A, int lda, size_t strideA,
                                     const __bf16* __restrict__ Bt, int ldb, size_t strideB,
                                     float* __restrict__ C, int ldc, size_t strideC,
                                     int Ntiles2, int tilesPerBatch, int K) {
    int wave = blockIdx.x * (blockDim.x >> 5) + (threadIdx.x >> 5);
    int lane = threadIdx.x & 31;
    int bidx = wave / tilesPerBatch;
    int t    = wave % tilesPerBatch;
    int mt = t / Ntiles2, nt = t % Ntiles2;
    int m0 = mt * 16 * MB, n0 = nt * 16 * NB;
    int hl = lane >> 4, l16 = lane & 15;
    const __bf16* Ab  = A  + (size_t)bidx * strideA;
    const __bf16* Btb = Bt + (size_t)bidx * strideB;
    float*        Cb  = C  + (size_t)bidx * strideC;
    v8f c[MB][NB];
#pragma unroll
    for (int i = 0; i < MB; ++i)
#pragma unroll
        for (int j = 0; j < NB; ++j) c[i][j] = (v8f){};
    for (int kb = 0; kb < K; kb += 32) {
        v16bf a[MB], b[NB];
#pragma unroll
        for (int i = 0; i < MB; ++i) a[i] = load_a_frag(Ab, lda, m0 + i * 16 + l16, kb, hl);
#pragma unroll
        for (int j = 0; j < NB; ++j) b[j] = load_b_frag(Btb, ldb, n0 + j * 16 + l16, kb, hl);
#pragma unroll
        for (int i = 0; i < MB; ++i)
#pragma unroll
            for (int j = 0; j < NB; ++j)
                c[i][j] = __builtin_amdgcn_wmma_f32_16x16x32_bf16(
                    false, a[i], false, b[j], (short)0, c[i][j], false, false);
    }
#pragma unroll
    for (int i = 0; i < MB; ++i)
#pragma unroll
        for (int j = 0; j < NB; ++j) {
            int n = n0 + j * 16 + l16;
#pragma unroll
            for (int r = 0; r < 8; ++r) {
                int row = m0 + i * 16 + r + hl * 8;
                Cb[(size_t)row * ldc + n] = c[i][j][r];
            }
        }
}

// ---------------------------------------------------------------------------
// Masked softmax over S2: one wave per (b,s1) row, 16 elems/lane.
// Faithful: masked positions get score -1e-5 BEFORE softmax.
// ---------------------------------------------------------------------------
__global__ void softmax_mask_kernel(const float* __restrict__ S,
                                    const int* __restrict__ x2_mask,
                                    __bf16* __restrict__ P) {
    int wave = blockIdx.x * (blockDim.x >> 5) + (threadIdx.x >> 5);
    int lane = threadIdx.x & 31;
    int b = wave >> 9;                      // / S1_
    const float* row = S + (size_t)wave * S2_;
    const int*   m   = x2_mask + (size_t)b * S2_;
    float vals[16];
    float mx = -INFINITY;
#pragma unroll
    for (int i = 0; i < 16; ++i) {
        int t = lane + 32 * i;
        float x = row[t];
        if (m[t] == 0) x = -1e-5f;
        vals[i] = x;
        mx = fmaxf(mx, x);
    }
#pragma unroll
    for (int off = 16; off >= 1; off >>= 1) mx = fmaxf(mx, __shfl_xor(mx, off, 32));
    float sum = 0.0f;
#pragma unroll
    for (int i = 0; i < 16; ++i) { vals[i] = __expf(vals[i] - mx); sum += vals[i]; }
#pragma unroll
    for (int off = 16; off >= 1; off >>= 1) sum += __shfl_xor(sum, off, 32);
    float inv = 1.0f / sum;
    __bf16* prow = P + (size_t)wave * S2_;
#pragma unroll
    for (int i = 0; i < 16; ++i) prow[lane + 32 * i] = (__bf16)(vals[i] * inv);
}

// ---------------------------------------------------------------------------
// Masked max-pool over S1: padded rows contribute -1e5 (faithful).
// ---------------------------------------------------------------------------
__global__ void maxpool_kernel(const float* __restrict__ v, int ldv,
                               const int* __restrict__ x1_mask,
                               float* __restrict__ out, int D, int ldo) {
    int idx = blockIdx.x * blockDim.x + threadIdx.x;
    if (idx >= B_ * D) return;
    int b = idx / D, d = idx % D;
    const float* vb = v + (size_t)b * S1_ * ldv + d;
    const int*   m  = x1_mask + (size_t)b * S1_;
    float mx = -INFINITY;
    for (int s = 0; s < S1_; ++s) {
        float x = (m[s] == 0) ? -1e5f : vb[(size_t)s * ldv];
        mx = fmaxf(mx, x);
    }
    out[(size_t)b * ldo + d] = mx;
}

// ---------------------------------------------------------------------------
// x12_wv [B,200] @ lin_wv_w [200,768] + b  ->  [B,768]  (tiny, fp32 VALU)
// ---------------------------------------------------------------------------
__global__ void wv_linear_kernel(const float* __restrict__ x12wv,
                                 const float* __restrict__ w,
                                 const float* __restrict__ bvec,
                                 float* __restrict__ out) {
    int idx = blockIdx.x * blockDim.x + threadIdx.x;
    if (idx >= B_ * H_) return;
    int b = idx / H_, h = idx % H_;
    float acc = bvec[h];
    for (int d = 0; d < DW_; ++d)
        acc += x12wv[(size_t)b * DWP + d] * w[(size_t)d * H_ + h];
    out[idx] = acc;
}

// ---------------------------------------------------------------------------
// o = sigmoid(concat(x1_h,x2_h,x12,x12_wv_h) @ lin_w + lin_b)
// ---------------------------------------------------------------------------
__global__ void final_kernel(const float* __restrict__ x1_h, const float* __restrict__ x2_h,
                             const float* __restrict__ x12, const float* __restrict__ x12wvh,
                             const float* __restrict__ lw, const float* __restrict__ lb,
                             float* __restrict__ o) {
    int b = blockIdx.x * blockDim.x + threadIdx.x;
    if (b >= B_) return;
    float acc = lb[0];
    for (int h = 0; h < H_; ++h) acc += x1_h[(size_t)b * H_ + h]   * lw[h];
    for (int h = 0; h < H_; ++h) acc += x2_h[(size_t)b * H_ + h]   * lw[H_ + h];
    for (int h = 0; h < H_; ++h) acc += x12[(size_t)b * H_ + h]    * lw[2 * H_ + h];
    for (int h = 0; h < H_; ++h) acc += x12wvh[(size_t)b * H_ + h] * lw[3 * H_ + h];
    o[b] = 1.0f / (1.0f + __expf(-acc));
}

__global__ void masks_kernel(const int* __restrict__ m1, const int* __restrict__ m2,
                             int* __restrict__ o1, int* __restrict__ o2) {
    int idx = blockIdx.x * blockDim.x + threadIdx.x;
    if (idx >= B_ * S1_) return;
    o1[idx] = 1 - m1[idx];
    o2[idx] = 1 - m2[idx];
}

// ---------------------------------------------------------------------------
// Launch
// ---------------------------------------------------------------------------
extern "C" void kernel_launch(void* const* d_in, const int* in_sizes, int n_in,
                              void* d_out, int out_size, void* d_ws, size_t ws_size,
                              hipStream_t stream) {
    (void)in_sizes; (void)n_in; (void)out_size; (void)ws_size;
    const float* x1       = (const float*)d_in[0];
    const float* x1_h     = (const float*)d_in[1];
    const int*   x1_mask  = (const int*)  d_in[2];
    const float* y        = (const float*)d_in[3];
    const float* x2       = (const float*)d_in[4];
    const float* x2_h     = (const float*)d_in[5];
    const int*   x2_mask  = (const int*)  d_in[6];
    const float* tt       = (const float*)d_in[7];
    const float* tt2      = (const float*)d_in[8];
    const float* W_bert   = (const float*)d_in[9];
    const float* W_wv     = (const float*)d_in[10];
    const float* lin_wv_w = (const float*)d_in[11];
    const float* lin_wv_b = (const float*)d_in[12];
    const float* lin_w    = (const float*)d_in[13];
    const float* lin_b    = (const float*)d_in[14];

    char* ws = (char*)d_ws;
    __bf16* A1bf  = (__bf16*)(ws + OFF_A1);
    __bf16* K1bf  = (__bf16*)(ws + OFF_K1);
    __bf16* X2T   = (__bf16*)(ws + OFF_X2T);
    __bf16* A2bf  = (__bf16*)(ws + OFF_A2);
    __bf16* K2bf  = (__bf16*)(ws + OFF_K2);
    __bf16* T2T   = (__bf16*)(ws + OFF_T2T);
    __bf16* WBT   = (__bf16*)(ws + OFF_WBT);
    float*  WLB   = (float*) (ws + OFF_WLB);
    __bf16* WWT   = (__bf16*)(ws + OFF_WWT);
    float*  WLW   = (float*) (ws + OFF_WLW);
    __bf16* Q1    = (__bf16*)(ws + OFF_Q1);
    __bf16* Q2    = (__bf16*)(ws + OFF_Q2);
    float*  Sbuf  = (float*) (ws + OFF_S);
    __bf16* Pbuf  = (__bf16*)(ws + OFF_P);
    float*  Vbuf  = (float*) (ws + OFF_V);
    float*  x12   = (float*) (ws + OFF_X12);
    float*  x12wv = (float*) (ws + OFF_X12W);
    float*  x12wvh= (float*) (ws + OFF_X12WH);

    const int TB = 256;           // 8 waves/block
    const long rowsX = (long)B_ * S1_;
    constexpr int MB = 4, NB = 2; // 64x32 macro-tile per wave

    // -- prep: bf16 packs / transposes -------------------------------------
    pack_bf16_kernel<<<(rowsX * H_  + TB - 1) / TB, TB, 0, stream>>>(x1,  A1bf, rowsX, H_,  H_);
    pack_bf16_kernel<<<(rowsX * H_  + TB - 1) / TB, TB, 0, stream>>>(x2,  K1bf, rowsX, H_,  H_);
    pack_bf16_kernel<<<(rowsX * DWP + TB - 1) / TB, TB, 0, stream>>>(tt,  A2bf, rowsX, DW_, DWP);
    pack_bf16_kernel<<<(rowsX * DWP + TB - 1) / TB, TB, 0, stream>>>(tt2, K2bf, rowsX, DW_, DWP);
    transpose_conv_kernel<<<((long)B_ * H_  * S2_ + TB - 1) / TB, TB, 0, stream>>>(x2,  X2T, S2_, H_,  H_);
    transpose_conv_kernel<<<((long)B_ * DWP * S2_ + TB - 1) / TB, TB, 0, stream>>>(tt2, T2T, S2_, DW_, DWP);
    prep_w_kernel<<<((long)H_  * H_  + TB - 1) / TB, TB, 0, stream>>>(W_bert, WBT, WLB, H_,  H_,  H_,  H_);
    prep_w_kernel<<<((long)DWP * DWP + TB - 1) / TB, TB, 0, stream>>>(W_wv,   WWT, WLW, DW_, DW_, DWP, DWP);

    // -- Q GEMMs (blocked WMMA bf16, fused rank-1 +y*W[last]) --------------
    {   // bert: M=32768, N=768, K=768 -> (512 x 24) macro-tiles
        int waves = (int)(rowsX / (16 * MB)) * (H_ / (16 * NB));
        wmma_gemm_q_kernel<MB, NB><<<waves / 8, TB, 0, stream>>>(
            A1bf, H_, WBT, H_, Q1, H_, y, WLB, H_ / (16 * NB), H_);
    }
    {   // wv: M=32768, N=224, K=224 -> (512 x 7) macro-tiles
        int waves = (int)(rowsX / (16 * MB)) * (DWP / (16 * NB));
        wmma_gemm_q_kernel<MB, NB><<<waves / 8, TB, 0, stream>>>(
            A2bf, DWP, WWT, DWP, Q2, DWP, y, WLW, DWP / (16 * NB), DWP);
    }

    // -- bert branch -------------------------------------------------------
    {   // scores: per batch (8 x 16) macro-tiles, K=768; Bt = x2 rows (Keys)
        int tpb = (S1_ / (16 * MB)) * (S2_ / (16 * NB));
        wmma_gemm_f32_kernel<MB, NB><<<(B_ * tpb) / 8, TB, 0, stream>>>(
            Q1, H_, (size_t)S1_ * H_, K1bf, H_, (size_t)S2_ * H_,
            Sbuf, S2_, (size_t)S1_ * S2_, S2_ / (16 * NB), tpb, H_);
    }
    softmax_mask_kernel<<<(B_ * S1_) / 8, TB, 0, stream>>>(Sbuf, x2_mask, Pbuf);
    {   // v = P @ x2: K=512, N=768; Bt = x2^T
        int tpb = (S1_ / (16 * MB)) * (H_ / (16 * NB));
        wmma_gemm_f32_kernel<MB, NB><<<(B_ * tpb) / 8, TB, 0, stream>>>(
            Pbuf, S2_, (size_t)S1_ * S2_, X2T, S2_, (size_t)H_ * S2_,
            Vbuf, H_, (size_t)S1_ * H_, H_ / (16 * NB), tpb, S2_);
    }
    maxpool_kernel<<<(B_ * H_ + TB - 1) / TB, TB, 0, stream>>>(Vbuf, H_, x1_mask, x12, H_, H_);

    // -- wv branch (reuses Sbuf/Pbuf/Vbuf) ---------------------------------
    {   // scores: K=224 (zero-padded); Bt = tt2 rows
        int tpb = (S1_ / (16 * MB)) * (S2_ / (16 * NB));
        wmma_gemm_f32_kernel<MB, NB><<<(B_ * tpb) / 8, TB, 0, stream>>>(
            Q2, DWP, (size_t)S1_ * DWP, K2bf, DWP, (size_t)S2_ * DWP,
            Sbuf, S2_, (size_t)S1_ * S2_, S2_ / (16 * NB), tpb, DWP);
    }
    softmax_mask_kernel<<<(B_ * S1_) / 8, TB, 0, stream>>>(Sbuf, x2_mask, Pbuf);
    {   // v = P @ tt2: K=512, N=224 (cols 200..223 come from zero rows of T2T)
        int tpb = (S1_ / (16 * MB)) * (DWP / (16 * NB));
        wmma_gemm_f32_kernel<MB, NB><<<(B_ * tpb) / 8, TB, 0, stream>>>(
            Pbuf, S2_, (size_t)S1_ * S2_, T2T, S2_, (size_t)DWP * S2_,
            Vbuf, DWP, (size_t)S1_ * DWP, DWP / (16 * NB), tpb, S2_);
    }
    maxpool_kernel<<<(B_ * DW_ + TB - 1) / TB, TB, 0, stream>>>(Vbuf, DWP, x1_mask, x12wv, DW_, DWP);

    // -- tails -------------------------------------------------------------
    wv_linear_kernel<<<(B_ * H_ + TB - 1) / TB, TB, 0, stream>>>(x12wv, lin_wv_w, lin_wv_b, x12wvh);
    final_kernel<<<1, 64, 0, stream>>>(x1_h, x2_h, x12, x12wvh, lin_w, lin_b, (float*)d_out);
    masks_kernel<<<(B_ * S1_ + TB - 1) / TB, TB, 0, stream>>>(
        x1_mask, x2_mask, (int*)d_out + B_, (int*)d_out + B_ + B_ * S1_);
}
